// SimpleBiLSTMSentenceEncoder_79328045957318
// MI455X (gfx1250) — compile-verified, matrix operand order checked by reading, not run on hardware
//
#include <hip/hip_runtime.h>
#include <hip/hip_bf16.h>
#include <stdint.h>

// ---------------- types ----------------
typedef __bf16 bf16;
typedef __attribute__((ext_vector_type(16))) bf16  v16bf;
typedef __attribute__((ext_vector_type(8)))  bf16  v8bf;
typedef __attribute__((ext_vector_type(8)))  float v8f;

#define HID   2048
#define G4H   8192       // 4*HID
#define EMBR  300
#define EMBP  320        // padded to multiple of 32
#define BSZ   64
#define TLEN  128

#define KCH   256        // K-chunk staged in LDS per iteration
#define LROW  264        // LDS row stride in elements (256 + 8 pad, keeps 16B align)

// workspace layout (byte offsets, all 256-aligned)
#define OFF_WHH   ((size_t)0)                         // bf16 [2][8192][2048]  64 MB
#define OFF_WIH   ((size_t)67108864)                  // bf16 [2][8192][320]   10 MB
#define OFF_X     ((size_t)77594624)                  // bf16 [2][128][64][320] 10 MB
#define OFF_BIAS  ((size_t)88080384)                  // f32  [2][8192]
#define OFF_H     ((size_t)88145920)                  // f32  [2][64][2048]
#define OFF_C     ((size_t)89194496)                  // f32  [2][64][2048]
#define OFF_HBF   ((size_t)90243072)                  // bf16 [2][64][2048]
#define OFF_GATES ((size_t)90767360)                  // f32  [2][64][8192]

__device__ __forceinline__ v16bf pack16(v8bf a, v8bf b) {
    return __builtin_shufflevector(a, b, 0,1,2,3,4,5,6,7,8,9,10,11,12,13,14,15);
}

// ---------------- preprocessing ----------------
__global__ void convert_pad_kernel(const float* __restrict__ in, bf16* __restrict__ out,
                                   int rows, int kin, int kout) {
    int idx = blockIdx.x * blockDim.x + threadIdx.x;
    if (idx >= rows * kout) return;
    int k = idx % kout;
    int r = idx / kout;
    float v = (k < kin) ? in[(size_t)r * kin + k] : 0.0f;
    out[idx] = (bf16)v;
}

__global__ void bias_sum_kernel(const float* __restrict__ a, const float* __restrict__ b,
                                float* __restrict__ out) {
    int i = blockIdx.x * blockDim.x + threadIdx.x;
    if (i < G4H) out[i] = a[i] + b[i];
}

// X[dir][t][b][k] = emb_dir[ tokens[b][ dir ? T-1-t : t ] ][k]  (bf16, zero-padded K)
__global__ void embed_gather_kernel(const int* __restrict__ tokens,
                                    const float* __restrict__ emb_f,
                                    const float* __restrict__ emb_b,
                                    bf16* __restrict__ X) {
    int idx = blockIdx.x * blockDim.x + threadIdx.x;
    if (idx >= 2 * TLEN * BSZ * EMBP) return;
    int k = idx % EMBP;
    int r = idx / EMBP;
    int b = r % BSZ; r /= BSZ;
    int t = r % TLEN;
    int dir = r / TLEN;
    int tt  = dir ? (TLEN - 1 - t) : t;
    int tok = tokens[b * TLEN + tt];
    const float* emb = dir ? emb_b : emb_f;
    float v = (k < EMBR) ? emb[(size_t)tok * EMBR + k] : 0.0f;
    X[idx] = (bf16)v;
}

__global__ void zero_u32_kernel(uint32_t* __restrict__ p, int n) {
    int i = blockIdx.x * blockDim.x + threadIdx.x;
    if (i < n) p[i] = 0u;
}

// ---------------- recurrent gates GEMM (WMMA bf16 -> f32, LDS-staged A) ----------------
// gates[dir][m][n] = sum_k h[dir][m][k]*Whh[dir][n][k] + sum_k X[dir][t][m][k]*Wih[dir][n][k]
// block = 256 threads (8 waves). Wave w owns N-tile n0 = bx*128 + w*16, all 4 M-tiles.
// h is staged into LDS in 64xKCH double-buffered chunks via async global->LDS copies.
__global__ __launch_bounds__(256)
void gates_gemm_kernel(const bf16* __restrict__ Whh,   // [2][8192][2048]
                       const bf16* __restrict__ Wih,   // [2][8192][320]
                       const bf16* __restrict__ Xall,  // [2][128][64][320]
                       const bf16* __restrict__ hbf,   // [2][64][2048]
                       float* __restrict__ gates,      // [2][64][8192]
                       int t) {
    __shared__ bf16 Abuf[2][BSZ][LROW];   // ~66 KB

    const int dir  = blockIdx.y;
    const int tid  = threadIdx.x;
    const int wave = tid >> 5;
    const int lane = tid & 31;
    const int half = lane >> 4;       // K-half select per 16-bit WMMA layout
    const int l16  = lane & 15;
    const int n0   = blockIdx.x * 128 + wave * 16;

    const bf16* hsrc = hbf + (size_t)dir * BSZ * HID;
    const bf16* Bw   = Whh + (size_t)dir * G4H * HID + (size_t)(n0 + l16) * HID;

    v8f acc0 = {}, acc1 = {}, acc2 = {}, acc3 = {};

    // async-stage one 64xKCH chunk of h into Abuf[buf] (8 b128 transfers per thread)
    auto stage = [&](int buf, int kbase) {
        #pragma unroll
        for (int i = 0; i < 8; ++i) {
            int linear = tid + 256 * i;            // 0..2047 (64 rows x 32 16B-units)
            int row  = linear >> 5;
            int c16  = linear & 31;
            const bf16* g = hsrc + (size_t)row * HID + kbase + c16 * 8;
            uint32_t l = (uint32_t)(uintptr_t)(&Abuf[buf][row][c16 * 8]);
            asm volatile("global_load_async_to_lds_b128 %0, %1, off"
                         :: "v"(l), "v"(g) : "memory");
        }
    };

    // ---- h @ Whh^T : K = 2048, in 8 chunks of KCH ----
    stage(0, 0);
    for (int kc = 0; kc < HID / KCH; ++kc) {
        const int buf = kc & 1;
        if (kc + 1 < HID / KCH) {
            stage(buf ^ 1, (kc + 1) * KCH);
            asm volatile("s_wait_asynccnt 0x8" ::: "memory");
        } else {
            asm volatile("s_wait_asynccnt 0x0" ::: "memory");
        }
        __syncthreads();   // all waves' staged data visible

        const bf16* Brow = Bw + kc * KCH;
        const bf16* L0   = &Abuf[buf][0][0];
        #pragma unroll
        for (int kk = 0; kk < KCH; kk += 32) {
            const int kb = kk + half * 8;
            v16bf bfr = pack16(*(const v8bf*)(Brow + kb), *(const v8bf*)(Brow + kb + 16));
            const bf16* a0 = L0 + (size_t)l16 * LROW + kb;
            v16bf a;
            a = pack16(*(const v8bf*)(a0),             *(const v8bf*)(a0 + 16));
            acc0 = __builtin_amdgcn_wmma_f32_16x16x32_bf16(false, a, false, bfr, (short)0, acc0, false, false);
            a = pack16(*(const v8bf*)(a0 + 16*LROW),   *(const v8bf*)(a0 + 16*LROW + 16));
            acc1 = __builtin_amdgcn_wmma_f32_16x16x32_bf16(false, a, false, bfr, (short)0, acc1, false, false);
            a = pack16(*(const v8bf*)(a0 + 32*LROW),   *(const v8bf*)(a0 + 32*LROW + 16));
            acc2 = __builtin_amdgcn_wmma_f32_16x16x32_bf16(false, a, false, bfr, (short)0, acc2, false, false);
            a = pack16(*(const v8bf*)(a0 + 48*LROW),   *(const v8bf*)(a0 + 48*LROW + 16));
            acc3 = __builtin_amdgcn_wmma_f32_16x16x32_bf16(false, a, false, bfr, (short)0, acc3, false, false);
        }
        __syncthreads();   // all waves done reading buf before it is overwritten
    }

    // ---- x_t @ Wih^T : K = 320 (padded), direct global loads ----
    {
        const bf16* Bx = Wih + (size_t)dir * G4H * EMBP + (size_t)(n0 + l16) * EMBP;
        const bf16* A0 = Xall + (((size_t)dir * TLEN + t) * BSZ) * EMBP + (size_t)(l16) * EMBP;
        for (int k = 0; k < EMBP; k += 32) {
            const int kb = k + half * 8;
            v16bf bfr = pack16(*(const v8bf*)(Bx + kb), *(const v8bf*)(Bx + kb + 16));
            v16bf a;
            a = pack16(*(const v8bf*)(A0 + kb),               *(const v8bf*)(A0 + kb + 16));
            acc0 = __builtin_amdgcn_wmma_f32_16x16x32_bf16(false, a, false, bfr, (short)0, acc0, false, false);
            a = pack16(*(const v8bf*)(A0 + 16*EMBP + kb),     *(const v8bf*)(A0 + 16*EMBP + kb + 16));
            acc1 = __builtin_amdgcn_wmma_f32_16x16x32_bf16(false, a, false, bfr, (short)0, acc1, false, false);
            a = pack16(*(const v8bf*)(A0 + 32*EMBP + kb),     *(const v8bf*)(A0 + 32*EMBP + kb + 16));
            acc2 = __builtin_amdgcn_wmma_f32_16x16x32_bf16(false, a, false, bfr, (short)0, acc2, false, false);
            a = pack16(*(const v8bf*)(A0 + 48*EMBP + kb),     *(const v8bf*)(A0 + 48*EMBP + kb + 16));
            acc3 = __builtin_amdgcn_wmma_f32_16x16x32_bf16(false, a, false, bfr, (short)0, acc3, false, false);
        }
    }

    // ---- store D: lane l16 -> column n0+l16; VGPR r -> row r + 8*half (+16*mt) ----
    float* gout = gates + (size_t)dir * BSZ * G4H + (n0 + l16);
    const int mrow = half * 8;
    #pragma unroll
    for (int r = 0; r < 8; ++r) {
        gout[(size_t)(      mrow + r) * G4H] = acc0[r];
        gout[(size_t)(16  + mrow + r) * G4H] = acc1[r];
        gout[(size_t)(32  + mrow + r) * G4H] = acc2[r];
        gout[(size_t)(48  + mrow + r) * G4H] = acc3[r];
    }
}

// ---------------- LSTM cell elementwise update ----------------
__device__ __forceinline__ float sigf(float x) { return 1.0f / (1.0f + __expf(-x)); }

__global__ void lstm_cell_kernel(const float* __restrict__ gates, const float* __restrict__ bias,
                                 float* __restrict__ h, float* __restrict__ c,
                                 bf16* __restrict__ hbf) {
    int idx = blockIdx.x * blockDim.x + threadIdx.x;     // over 2*64*2048
    if (idx >= 2 * BSZ * HID) return;
    int dir = idx / (BSZ * HID);
    int rem = idx % (BSZ * HID);
    int b = rem / HID, n = rem % HID;
    const float* g  = gates + ((size_t)dir * BSZ + b) * G4H;
    const float* bs = bias + (size_t)dir * G4H;
    float iv = sigf (g[n]            + bs[n]);
    float fv = sigf (g[HID + n]      + bs[HID + n]);
    float gv = tanhf(g[2*HID + n]    + bs[2*HID + n]);
    float ov = sigf (g[3*HID + n]    + bs[3*HID + n]);
    size_t off = ((size_t)dir * BSZ + b) * HID + n;
    float cn = fv * c[off] + iv * gv;
    c[off] = cn;
    float hn = ov * tanhf(cn);
    h[off] = hn;
    hbf[off] = (bf16)hn;
}

// out[b][0:2048] = h_fwd[b], out[b][2048:4096] = h_bwd[b]
__global__ void write_out_kernel(const float* __restrict__ h, float* __restrict__ out) {
    int idx = blockIdx.x * blockDim.x + threadIdx.x;     // over 64*4096
    if (idx >= BSZ * 2 * HID) return;
    int b = idx / (2 * HID);
    int col = idx % (2 * HID);
    int dir = col / HID;
    int n = col % HID;
    out[idx] = h[((size_t)dir * BSZ + b) * HID + n];
}

// ---------------- launcher ----------------
extern "C" void kernel_launch(void* const* d_in, const int* in_sizes, int n_in,
                              void* d_out, int out_size, void* d_ws, size_t ws_size,
                              hipStream_t stream) {
    (void)in_sizes; (void)n_in; (void)out_size; (void)ws_size;
    const int*   tokens = (const int*)  d_in[0];
    const float* emb_f  = (const float*)d_in[1];
    const float* Wih_f  = (const float*)d_in[2];
    const float* Whh_f  = (const float*)d_in[3];
    const float* bih_f  = (const float*)d_in[4];
    const float* bhh_f  = (const float*)d_in[5];
    const float* emb_b  = (const float*)d_in[6];
    const float* Wih_b  = (const float*)d_in[7];
    const float* Whh_b  = (const float*)d_in[8];
    const float* bih_b  = (const float*)d_in[9];
    const float* bhh_b  = (const float*)d_in[10];

    char* ws = (char*)d_ws;
    bf16*  Whh   = (bf16*) (ws + OFF_WHH);
    bf16*  Wih   = (bf16*) (ws + OFF_WIH);
    bf16*  X     = (bf16*) (ws + OFF_X);
    float* bias  = (float*)(ws + OFF_BIAS);
    float* h     = (float*)(ws + OFF_H);
    float* c     = (float*)(ws + OFF_C);
    bf16*  hbf   = (bf16*) (ws + OFF_HBF);
    float* gates = (float*)(ws + OFF_GATES);

    // 1) weights -> bf16 (Wih padded 300 -> 320)
    {
        int nWhh = G4H * HID, nWih = G4H * EMBP;
        convert_pad_kernel<<<(nWhh + 255) / 256, 256, 0, stream>>>(Whh_f, Whh,                G4H, HID,  HID);
        convert_pad_kernel<<<(nWhh + 255) / 256, 256, 0, stream>>>(Whh_b, Whh + (size_t)nWhh, G4H, HID,  HID);
        convert_pad_kernel<<<(nWih + 255) / 256, 256, 0, stream>>>(Wih_f, Wih,                G4H, EMBR, EMBP);
        convert_pad_kernel<<<(nWih + 255) / 256, 256, 0, stream>>>(Wih_b, Wih + (size_t)nWih, G4H, EMBR, EMBP);
    }
    // 2) bias sums
    bias_sum_kernel<<<(G4H + 255) / 256, 256, 0, stream>>>(bih_f, bhh_f, bias);
    bias_sum_kernel<<<(G4H + 255) / 256, 256, 0, stream>>>(bih_b, bhh_b, bias + G4H);
    // 3) embedding gather (both directions; backward time-reversed here)
    {
        int nX = 2 * TLEN * BSZ * EMBP;
        embed_gather_kernel<<<(nX + 255) / 256, 256, 0, stream>>>(tokens, emb_f, emb_b, X);
    }
    // 4) zero h, c (f32) and h bf16 — contiguous region starting at OFF_H
    {
        int nwords = (int)((OFF_HBF + (size_t)2 * BSZ * HID * sizeof(bf16) - OFF_H) / 4);
        zero_u32_kernel<<<(nwords + 255) / 256, 256, 0, stream>>>((uint32_t*)(ws + OFF_H), nwords);
    }
    // 5) sequential scan: 128 steps, both directions per launch
    dim3 ggrid(G4H / 128, 2);
    int cellN = 2 * BSZ * HID;
    for (int t = 0; t < TLEN; ++t) {
        gates_gemm_kernel<<<ggrid, 256, 0, stream>>>(Whh, Wih, X, hbf, gates, t);
        lstm_cell_kernel<<<(cellN + 255) / 256, 256, 0, stream>>>(gates, bias, h, c, hbf);
    }
    // 6) concat [h_fwd, h_bwd] -> d_out
    {
        int nOut = BSZ * 2 * HID;
        write_out_kernel<<<(nOut + 255) / 256, 256, 0, stream>>>(h, (float*)d_out);
    }
}
